// Attention_82162724373196
// MI455X (gfx1250) — matrix operand order repaired and checked by reading, not compiled, not run on hardware
//
#include <hip/hip_runtime.h>

// ---------------------------------------------------------------------------
// Multi-head attention w/ distance-bias MLP for MI455X (gfx1250, wave32).
// B=2, F=T=2048, D=1024, N=16 heads, H=64, K_mlp=16.
// All GEMM stages use v_wmma_f32_16x16x32_bf16 (f32 accum).
// Flash-attention structure: logits never hit HBM (would be 1.07 GB).
// Distance-MLP bias precomputed once (shared across the 16 heads).
// ---------------------------------------------------------------------------

#define BB   2
#define FF   2048
#define TT   2048
#define DD   1024
#define NHD  16
#define HH   64
#define KM   16

typedef __attribute__((ext_vector_type(16))) __bf16 bf16x16;
typedef __attribute__((ext_vector_type(8)))  __bf16 bf16x8;
typedef __attribute__((ext_vector_type(8)))  float  f32x8;

__device__ __forceinline__ f32x8 wmma_bf16(bf16x16 a, bf16x16 b, f32x8 c) {
  // (neg_a, A, neg_b, B, c_mod, C, reuse_a, reuse_b)
  return __builtin_amdgcn_wmma_f32_16x16x32_bf16(false, a, false, b, (short)0, c,
                                                 false, false);
}

__device__ __forceinline__ bf16x16 cat8(bf16x8 lo, bf16x8 hi) {
  return __builtin_shufflevector(lo, hi, 0, 1, 2, 3, 4, 5, 6, 7,
                                 8, 9, 10, 11, 12, 13, 14, 15);
}

__device__ __forceinline__ f32x8 zero8() {
  f32x8 z;
#pragma unroll
  for (int i = 0; i < 8; ++i) z[i] = 0.0f;
  return z;
}

// ---------------------------------------------------------------------------
// Pack a [K=1024][C=1024] row-major f32 weight matrix into bf16 WMMA
// B-fragment order: WP[((ct*32 + kt)*32 + lane)*16 + j] = W[k][c]
//   with k = kt*32 + (j<8 ? (lane/16)*8 + j : 16 + (lane/16)*8 + j-8), c = ct*16 + (lane%16).
// GEMM inner loop then does one contiguous 32-byte v16bf load per lane.
// ---------------------------------------------------------------------------
__global__ __launch_bounds__(256) void pack_w_kernel(const float* __restrict__ W,
                                                     __bf16* __restrict__ WP) {
  int idx = blockIdx.x * 256 + threadIdx.x;  // total = 1024*1024
  if (idx >= DD * DD) return;
  int j    = idx & 15;
  int lane = (idx >> 4) & 31;
  int tmp  = idx >> 9;
  int kt   = tmp & 31;  // 1024/32 = 32 k-tiles
  int ct   = tmp >> 5;  // 1024/16 = 64 col-tiles
  int k = kt * 32 + (j < 8 ? ((lane >> 4) << 3) + j : 16 + ((lane >> 4) << 3) + (j - 8));
  int c = ct * 16 + (lane & 15);
  WP[idx] = (__bf16)W[(size_t)k * DD + c];
}

// ---------------------------------------------------------------------------
// bias_total[b,f,t] = bias[b,0,f,t] + (relu(dist*Wb1 + bb1) @ Wb2 + bb2)
// Done once; shared by all 16 heads (16x VALU saving vs inline).
// ---------------------------------------------------------------------------
__global__ __launch_bounds__(256) void bias_kernel(const float* __restrict__ dist,
                                                   const float* __restrict__ biasIn,
                                                   const float* __restrict__ Wb1,
                                                   const float* __restrict__ bb1,
                                                   const float* __restrict__ Wb2,
                                                   const float* __restrict__ bb2,
                                                   float* __restrict__ biasT) {
  int idx = blockIdx.x * 256 + threadIdx.x;  // B*F*T = 8,388,608
  if (idx >= BB * FF * TT) return;
  float d = dist[idx];
  float a = bb2[0] + biasIn[idx];
#pragma unroll
  for (int k = 0; k < KM; ++k)
    a += fmaxf(fmaf(d, Wb1[k], bb1[k]), 0.0f) * Wb2[k];
  biasT[idx] = a;
}

// ---------------------------------------------------------------------------
// Projection: Y = scale * (X[4096,1024] @ W[1024,1024]), stored as bf16 in
// head-major layout.  vtrans=0: Y[b][hd][row][h]  (Q and K)
//                     vtrans=1: Y[b][hd][h][row]  (V, T-major for B-fragments)
// Block = 128 (4 waves), each wave owns one 16x16 output tile.
// ---------------------------------------------------------------------------
__global__ __launch_bounds__(128) void proj_kernel(const float* __restrict__ X,
                                                   const __bf16* __restrict__ WP,
                                                   __bf16* __restrict__ Y,
                                                   float scale, int vtrans) {
  int lane  = threadIdx.x & 31;
  int wave  = threadIdx.x >> 5;
  int rt    = blockIdx.x;             // 0..255 row tiles
  int ct    = blockIdx.y * 4 + wave;  // 0..63 col tiles
  int m     = lane & 15;
  int khalf = lane >> 4;

  f32x8 acc = zero8();
  const float* xrow = X + (size_t)(rt * 16 + m) * DD;
  for (int kt = 0; kt < 32; ++kt) {
    // A-fragment: two contiguous 8-float runs per lane, cvt f32->bf16
    const float* xp = xrow + kt * 32;
    bf16x16 a;
#pragma unroll
    for (int j = 0; j < 8; ++j) {
      a[j]     = (__bf16)xp[khalf * 8 + j];
      a[j + 8] = (__bf16)xp[16 + khalf * 8 + j];
    }
    bf16x16 w = *(const bf16x16*)(WP + ((size_t)(ct * 32 + kt) * 32 + lane) * 16);
    acc = wmma_bf16(a, w, acc);
  }

  int c = ct * 16 + m;
  int hd = c >> 6, h = c & 63;
#pragma unroll
  for (int r = 0; r < 8; ++r) {
    int g  = rt * 16 + r + khalf * 8;  // global row in [0, 4096)
    int b  = g >> 11;
    int fr = g & 2047;
    float val = acc[r] * scale;
    size_t off = vtrans
        ? (((size_t)(b * NHD + hd) * HH + h) * TT + fr)
        : (((size_t)(b * NHD + hd) * FF + fr) * HH + h);
    Y[off] = (__bf16)val;
  }
}

// ---------------------------------------------------------------------------
// Flash attention per (f-tile of 16, head, batch); one wave32 per block.
// S-tile = 16x32 per step: 4 WMMA for S, 4 WMMA for P@V.  Online softmax
// stats kept per-lane for the 8 C-layout rows; row reductions via shfl_xor
// inside the 16-lane column groups.  P converted C->A layout via 1KB LDS.
// ---------------------------------------------------------------------------
__global__ __launch_bounds__(32) void flash_kernel(const __bf16* __restrict__ Qb,
                                                   const __bf16* __restrict__ Kb,
                                                   const __bf16* __restrict__ Vt,
                                                   const float* __restrict__ biasT,
                                                   __bf16* __restrict__ attnb) {
  int lane  = threadIdx.x;
  int ftile = blockIdx.x;  // 0..127
  int hd    = blockIdx.y;  // 0..15
  int b     = blockIdx.z;  // 0..1
  int fbase = ftile * 16;
  int m     = lane & 15;
  int khalf = lane >> 4;
  int rowoff = khalf * 8;

  const __bf16* Qh = Qb + (size_t)(b * NHD + hd) * FF * HH;
  const __bf16* Kh = Kb + (size_t)(b * NHD + hd) * TT * HH;
  const __bf16* Vh = Vt + (size_t)(b * NHD + hd) * HH * TT;

  // Q A-fragments for this 16-row tile (H=64 -> two K=32 chunks)
  const __bf16* qrow = Qh + (size_t)(fbase + m) * HH;
  bf16x16 q0 = cat8(*(const bf16x8*)(qrow + khalf * 16),
                    *(const bf16x8*)(qrow + 8 + khalf * 16));
  bf16x16 q1 = cat8(*(const bf16x8*)(qrow + 32 + khalf * 16),
                    *(const bf16x8*)(qrow + 40 + khalf * 16));

  float mrow[8], lrow[8];
  f32x8 acc[4];
#pragma unroll
  for (int r = 0; r < 8; ++r) { mrow[r] = -1e30f; lrow[r] = 0.0f; }
#pragma unroll
  for (int h4 = 0; h4 < 4; ++h4) acc[h4] = zero8();

  __shared__ __bf16 Plds[16 * 32];

  for (int tb = 0; tb < TT; tb += 32) {
    // ---- S = Q @ K^T for two 16-col subtiles -----------------------------
    const __bf16* krow0 = Kh + (size_t)(tb + m) * HH;
    const __bf16* krow1 = Kh + (size_t)(tb + 16 + m) * HH;
    bf16x16 k00 = *(const bf16x16*)(krow0 + khalf * 16);
    bf16x16 k01 = *(const bf16x16*)(krow0 + 32 + khalf * 16);
    bf16x16 k10 = *(const bf16x16*)(krow1 + khalf * 16);
    bf16x16 k11 = *(const bf16x16*)(krow1 + 32 + khalf * 16);
    f32x8 s0 = zero8(), s1 = zero8();
    s0 = wmma_bf16(q0, k00, s0);
    s0 = wmma_bf16(q1, k01, s0);
    s1 = wmma_bf16(q0, k10, s1);
    s1 = wmma_bf16(q1, k11, s1);

    if (tb + 32 < TT) {  // gfx1250 global_prefetch_b8 for next tiles
      __builtin_prefetch(krow0 + 32 * HH, 0, 1);
      __builtin_prefetch(krow1 + 32 * HH, 0, 1);
    }

    // ---- add precomputed (dist-MLP + input) bias -------------------------
#pragma unroll
    for (int r = 0; r < 8; ++r) {
      int f = fbase + r + rowoff;
      size_t rbase = ((size_t)b * FF + f) * TT + tb;
      s0[r] += biasT[rbase + m];
      s1[r] += biasT[rbase + 16 + m];
    }

    // ---- online softmax update -------------------------------------------
#pragma unroll
    for (int r = 0; r < 8; ++r) {
      float mt = fmaxf(s0[r], s1[r]);
      mt = fmaxf(mt, __shfl_xor(mt, 8));
      mt = fmaxf(mt, __shfl_xor(mt, 4));
      mt = fmaxf(mt, __shfl_xor(mt, 2));
      mt = fmaxf(mt, __shfl_xor(mt, 1));
      float mnew  = fmaxf(mrow[r], mt);
      float alpha = __expf(mrow[r] - mnew);
      float p0 = __expf(s0[r] - mnew);
      float p1 = __expf(s1[r] - mnew);
      float rs = p0 + p1;
      rs += __shfl_xor(rs, 8);
      rs += __shfl_xor(rs, 4);
      rs += __shfl_xor(rs, 2);
      rs += __shfl_xor(rs, 1);
      lrow[r] = lrow[r] * alpha + rs;
      mrow[r] = mnew;
#pragma unroll
      for (int h4 = 0; h4 < 4; ++h4) acc[h4][r] *= alpha;
      // park P (bf16) in LDS in row-major [m][t] for C->A relayout
      Plds[(r + rowoff) * 32 + m]      = (__bf16)p0;
      Plds[(r + rowoff) * 32 + 16 + m] = (__bf16)p1;
    }
    __syncthreads();

    // ---- O += P @ V ------------------------------------------------------
    const __bf16* prow = Plds + m * 32;
    bf16x16 pf = cat8(*(const bf16x8*)(prow + khalf * 16),
                      *(const bf16x8*)(prow + 8 + khalf * 16));
#pragma unroll
    for (int h4 = 0; h4 < 4; ++h4) {
      bf16x16 vf = *(const bf16x16*)(Vh + (size_t)(h4 * 16 + m) * TT + tb +
                                     khalf * 16);
      acc[h4] = wmma_bf16(pf, vf, acc[h4]);
    }
    __syncthreads();
  }

  // ---- normalize and write attn (bf16, row-major [b*F+f][n*H+h]) ---------
#pragma unroll
  for (int h4 = 0; h4 < 4; ++h4) {
#pragma unroll
    for (int r = 0; r < 8; ++r) {
      float o = acc[h4][r] / lrow[r];
      int f = fbase + r + rowoff;
      int c = hd * HH + h4 * 16 + m;
      attnb[((size_t)b * FF + f) * DD + c] = (__bf16)o;
    }
  }
}

// ---------------------------------------------------------------------------
// Output projection: out[4096,1024] = attnb[4096,1024] @ Wo[1024,1024]
// ---------------------------------------------------------------------------
__global__ __launch_bounds__(128) void outproj_kernel(const __bf16* __restrict__ A,
                                                      const __bf16* __restrict__ WP,
                                                      float* __restrict__ out) {
  int lane  = threadIdx.x & 31;
  int wave  = threadIdx.x >> 5;
  int rt    = blockIdx.x;
  int ct    = blockIdx.y * 4 + wave;
  int m     = lane & 15;
  int khalf = lane >> 4;

  f32x8 acc = zero8();
  const __bf16* arow = A + (size_t)(rt * 16 + m) * DD;
  for (int kt = 0; kt < 32; ++kt) {
    bf16x16 a = cat8(*(const bf16x8*)(arow + kt * 32 + khalf * 8),
                     *(const bf16x8*)(arow + kt * 32 + 16 + khalf * 8));
    bf16x16 w = *(const bf16x16*)(WP + ((size_t)(ct * 32 + kt) * 32 + lane) * 16);
    acc = wmma_bf16(a, w, acc);
  }
#pragma unroll
  for (int r = 0; r < 8; ++r)
    out[(size_t)(rt * 16 + r + khalf * 8) * DD + ct * 16 + m] = acc[r];
}

// ---------------------------------------------------------------------------
extern "C" void kernel_launch(void* const* d_in, const int* in_sizes, int n_in,
                              void* d_out, int out_size, void* d_ws, size_t ws_size,
                              hipStream_t stream) {
  (void)in_sizes; (void)n_in; (void)out_size; (void)ws_size;
  const float* q_in  = (const float*)d_in[0];   // [B,F,D]
  const float* s_in  = (const float*)d_in[1];   // [B,T,D]
  const float* dist  = (const float*)d_in[2];   // [B,F,T]
  const float* biasI = (const float*)d_in[3];   // [B,1,F,T]
  const float* Wq    = (const float*)d_in[4];
  const float* Wk    = (const float*)d_in[5];
  const float* Wv    = (const float*)d_in[6];
  const float* Wo    = (const float*)d_in[7];
  const float* Wb1   = (const float*)d_in[8];
  const float* bb1   = (const float*)d_in[9];
  const float* Wb2   = (const float*)d_in[10];
  const float* bb2   = (const float*)d_in[11];
  float* out = (float*)d_out;

  // workspace partition (bytes): 4x2MB packed W + 3x8MB QKV + 8MB attn + 32MB bias
  char* ws = (char*)d_ws;
  size_t off = 0;
  auto take = [&](size_t bytes) {
    void* p = ws + off;
    off += (bytes + 255) & ~(size_t)255;
    return p;
  };
  __bf16* WqP   = (__bf16*)take((size_t)DD * DD * 2);
  __bf16* WkP   = (__bf16*)take((size_t)DD * DD * 2);
  __bf16* WvP   = (__bf16*)take((size_t)DD * DD * 2);
  __bf16* WoP   = (__bf16*)take((size_t)DD * DD * 2);
  __bf16* Qbuf  = (__bf16*)take((size_t)BB * NHD * FF * HH * 2);
  __bf16* Kbuf  = (__bf16*)take((size_t)BB * NHD * TT * HH * 2);
  __bf16* Vbuf  = (__bf16*)take((size_t)BB * NHD * HH * TT * 2);
  __bf16* Attn  = (__bf16*)take((size_t)BB * FF * DD * 2);
  float*  biasT = (float*)take((size_t)BB * FF * TT * 4);

  // 1) pack weights into WMMA B-fragment order (bf16)
  pack_w_kernel<<<4096, 256, 0, stream>>>(Wq, WqP);
  pack_w_kernel<<<4096, 256, 0, stream>>>(Wk, WkP);
  pack_w_kernel<<<4096, 256, 0, stream>>>(Wv, WvP);
  pack_w_kernel<<<4096, 256, 0, stream>>>(Wo, WoP);

  // 2) fused distance-MLP + additive bias (once, shared across heads)
  bias_kernel<<<(BB * FF * TT + 255) / 256, 256, 0, stream>>>(
      dist, biasI, Wb1, bb1, Wb2, bb2, biasT);

  // 3) Q/K/V projections (Q pre-scaled by H^-0.5 = 0.125)
  dim3 pg(256, 16);
  proj_kernel<<<pg, 128, 0, stream>>>(q_in, WqP, Qbuf, 0.125f, 0);
  proj_kernel<<<pg, 128, 0, stream>>>(s_in, WkP, Kbuf, 1.0f, 0);
  proj_kernel<<<pg, 128, 0, stream>>>(s_in, WvP, Vbuf, 1.0f, 1);

  // 4) flash attention
  flash_kernel<<<dim3(FF / 16, NHD, BB), 32, 0, stream>>>(Qbuf, Kbuf, Vbuf,
                                                          biasT, Attn);

  // 5) output projection
  outproj_kernel<<<pg, 128, 0, stream>>>(Attn, WoP, out);
}
